// GroupedQueryAttention_39350490366253
// MI455X (gfx1250) — compile-verified, matrix operand order checked by reading, not run on hardware
//
#include <hip/hip_runtime.h>

typedef __bf16 v16bf __attribute__((ext_vector_type(16)));
typedef __bf16 v8bf  __attribute__((ext_vector_type(8)));
typedef float  v8f   __attribute__((ext_vector_type(8)));

union BFrag { v16bf v; v8bf h[2]; };

#define B_SZ   2
#define T_SEQ  2048
#define CMODEL 2048
#define DKV    512
#define HEADS  16
#define GROUPS 4
#define HD     128
#define MROWS  (B_SZ * T_SEQ)   // 4096

// ---------------- f32 -> bf16 staging (8 elems / thread) ----------------
__global__ __launch_bounds__(256)
void cvt_f32_to_bf16(const float* __restrict__ in, __bf16* __restrict__ out, int n8)
{
    int i = blockIdx.x * 256 + threadIdx.x;
    if (i < n8) {
        const float4* p = (const float4*)in + (size_t)i * 2;
        float4 a = p[0], b = p[1];
        v8bf o = { (__bf16)a.x, (__bf16)a.y, (__bf16)a.z, (__bf16)a.w,
                   (__bf16)b.x, (__bf16)b.y, (__bf16)b.z, (__bf16)b.w };
        *((v8bf*)out + i) = o;
    }
}

// ---------------- NT GEMM: C[M,N] = A[M,K] * B[N,K]^T, bf16 in/out ----------
// One wave computes a 64x64 tile (16 WMMA per K-step of 32).
__global__ __launch_bounds__(32)
void gemm_bf16_nt(const __bf16* __restrict__ A, const __bf16* __restrict__ B,
                  __bf16* __restrict__ C, int M, int N, int K, int transOut)
{
    const int lane  = threadIdx.x;
    const int r     = lane & 15;
    const int hh    = lane >> 4;
    const int mbase = blockIdx.x * 64;
    const int nbase = blockIdx.y * 64;

    const v8f zero = {};
    v8f acc[4][4];
#pragma unroll
    for (int i = 0; i < 4; ++i)
#pragma unroll
        for (int j = 0; j < 4; ++j) acc[i][j] = zero;

#pragma unroll 1
    for (int kb = 0; kb < K; kb += 32) {
        BFrag a[4];
#pragma unroll
        for (int i = 0; i < 4; ++i) {
            const __bf16* pa = A + (size_t)(mbase + i * 16 + r) * K + kb + hh * 8;
            a[i].h[0] = *(const v8bf*)pa;
            a[i].h[1] = *(const v8bf*)(pa + 16);
        }
#pragma unroll
        for (int j = 0; j < 4; ++j) {
            const __bf16* pb = B + (size_t)(nbase + j * 16 + r) * K + kb + hh * 8;
            BFrag bf;
            bf.h[0] = *(const v8bf*)pb;
            bf.h[1] = *(const v8bf*)(pb + 16);
#pragma unroll
            for (int i = 0; i < 4; ++i)
                acc[i][j] = __builtin_amdgcn_wmma_f32_16x16x32_bf16(
                    false, a[i].v, false, bf.v, (short)0, acc[i][j], false, false);
        }
    }

    if (transOut) {
#pragma unroll
        for (int i = 0; i < 4; ++i)
#pragma unroll
            for (int j = 0; j < 4; ++j)
#pragma unroll
                for (int e = 0; e < 8; ++e) {
                    int row = mbase + i * 16 + hh * 8 + e;
                    int col = nbase + j * 16 + r;
                    C[(size_t)col * M + row] = (__bf16)acc[i][j][e];
                }
    } else {
#pragma unroll
        for (int i = 0; i < 4; ++i)
#pragma unroll
            for (int j = 0; j < 4; ++j)
#pragma unroll
                for (int e = 0; e < 8; ++e) {
                    int row = mbase + i * 16 + hh * 8 + e;
                    int col = nbase + j * 16 + r;
                    C[(size_t)row * N + col] = (__bf16)acc[i][j][e];
                }
    }
}

// ---------------- Flash attention -------------------------------------------
// Workgroup = 4 waves = 64 query rows of one (b,head); K/V chunks staged in LDS.
// Q  : bf16 [B*T, 2048]  (head h at col h*128)
// Kb : bf16 [B*T, 512]   (group g at col g*128)
// Vt : bf16 [512, B*T]   (transposed: Vt[g*128+d, b*T + t])
// out: f32  [B, T, 2048]
__global__ __launch_bounds__(128)
void gqa_flash_attn(const __bf16* __restrict__ Q, const __bf16* __restrict__ Kb,
                    const __bf16* __restrict__ Vt, float* __restrict__ out)
{
    __shared__ __bf16 ksh[32 * 128];      // K chunk: [key 0..31][d 0..127]   8KB
    __shared__ __bf16 vsh[128 * 32];      // V chunk: [d 0..127][key 0..31]   8KB
    __shared__ __bf16 ptile[4][16 * 32];  // per-wave P bounce tiles          4KB

    const int tid   = threadIdx.x;
    const int wid   = tid >> 5;
    const int lane  = tid & 31;
    const int r     = lane & 15;
    const int hh    = lane >> 4;
    const int bh    = blockIdx.x;
    const int b     = bh >> 4;
    const int head  = bh & 15;
    const int g     = head >> 2;
    const int wgq   = blockIdx.y * 64;    // workgroup query base
    const int qbase = wgq + wid * 16;     // this wave's query base

    const float scale = 0.088388347648318447f;  // 1/sqrt(128)
    const float NEG   = -1.0e30f;

    const __bf16* Qp = Q  + (size_t)(b * T_SEQ) * CMODEL + head * HD;
    const __bf16* Kp = Kb + (size_t)(b * T_SEQ) * DKV    + g * HD;
    const __bf16* Vp = Vt + (size_t)(g * HD) * MROWS + (size_t)b * T_SEQ;

    // Q fragments: 16 rows x 128 features = 4 K-chunks of 32
    BFrag qf[4];
#pragma unroll
    for (int kk = 0; kk < 4; ++kk) {
        const __bf16* p = Qp + (size_t)(qbase + r) * CMODEL + kk * 32 + hh * 8;
        qf[kk].h[0] = *(const v8bf*)p;
        qf[kk].h[1] = *(const v8bf*)(p + 16);
    }

    const v8f zero = {};
    v8f O[8];
#pragma unroll
    for (int nt = 0; nt < 8; ++nt) O[nt] = zero;
    float mrow[8], lrow[8];
#pragma unroll
    for (int e = 0; e < 8; ++e) { mrow[e] = NEG; lrow[e] = 0.f; }

    // uniform trip count over the workgroup (last row = wgq+63)
    const int nchunk = (wgq + 63) / 32 + 1;
#pragma unroll 1
    for (int c = 0; c < nchunk; ++c) {
        const int kb0 = c * 32;

        // -- cooperative stage: K chunk (8KB) + V chunk (8KB) into LDS --
#pragma unroll
        for (int i = 0; i < 4; ++i) {
            int seg = tid + i * 128;          // 0..511 segments of 16B
            int row = seg >> 4;               // key 0..31
            int off = (seg & 15) * 8;         // d offset
            *(v8bf*)&ksh[row * 128 + off] =
                *(const v8bf*)(Kp + (size_t)(kb0 + row) * DKV + off);
        }
#pragma unroll
        for (int i = 0; i < 4; ++i) {
            int seg = tid + i * 128;          // 0..511
            int d   = seg >> 2;               // d 0..127
            int off = (seg & 3) * 8;          // key offset
            *(v8bf*)&vsh[d * 32 + off] =
                *(const v8bf*)(Vp + (size_t)d * MROWS + kb0 + off);
        }
        __syncthreads();

        // -- S = Q * K^T (two 16x16 tiles) --
        v8f s[2];
#pragma unroll
        for (int t = 0; t < 2; ++t) {
            s[t] = zero;
#pragma unroll
            for (int kk = 0; kk < 4; ++kk) {
                const __bf16* p = &ksh[(t * 16 + r) * 128 + kk * 32 + hh * 8];
                BFrag kf;
                kf.h[0] = *(const v8bf*)p;
                kf.h[1] = *(const v8bf*)(p + 16);
                s[t] = __builtin_amdgcn_wmma_f32_16x16x32_bf16(
                    false, qf[kk].v, false, kf.v, (short)0, s[t], false, false);
            }
        }

        // -- scale + causal mask (row = qbase+e+8*hh, col = kb0+t*16+r) --
#pragma unroll
        for (int t = 0; t < 2; ++t)
#pragma unroll
            for (int e = 0; e < 8; ++e) {
                int row = qbase + e + 8 * hh;
                int col = kb0 + t * 16 + r;
                float sv = s[t][e] * scale;
                s[t][e] = (col > row) ? NEG : sv;
            }

        // -- online softmax (reduce across the 16 lanes of each half) --
        float alpha[8];
#pragma unroll
        for (int e = 0; e < 8; ++e) {
            float mx = fmaxf(s[0][e], s[1][e]);
#pragma unroll
            for (int off = 1; off < 16; off <<= 1)
                mx = fmaxf(mx, __shfl_xor(mx, off, 32));
            float mnew = fmaxf(mrow[e], mx);
            float p0 = __expf(s[0][e] - mnew);
            float p1 = __expf(s[1][e] - mnew);
            float ps = p0 + p1;
#pragma unroll
            for (int off = 1; off < 16; off <<= 1)
                ps += __shfl_xor(ps, off, 32);
            alpha[e] = __expf(mrow[e] - mnew);
            lrow[e]  = lrow[e] * alpha[e] + ps;
            mrow[e]  = mnew;
            ptile[wid][(e + 8 * hh) * 32 + r]      = (__bf16)p0;
            ptile[wid][(e + 8 * hh) * 32 + 16 + r] = (__bf16)p1;
        }

        asm volatile("s_wait_dscnt 0x0" ::: "memory");   // same-wave LDS RAW

        BFrag pf;
        pf.h[0] = *(const v8bf*)&ptile[wid][r * 32 + hh * 8];
        pf.h[1] = *(const v8bf*)&ptile[wid][r * 32 + 16 + hh * 8];

        // -- O = diag(alpha)*O + P * V_chunk --
#pragma unroll
        for (int nt = 0; nt < 8; ++nt) {
#pragma unroll
            for (int e = 0; e < 8; ++e) O[nt][e] *= alpha[e];
            const __bf16* p = &vsh[(nt * 16 + r) * 32 + hh * 8];
            BFrag vf;
            vf.h[0] = *(const v8bf*)p;
            vf.h[1] = *(const v8bf*)(p + 16);
            O[nt] = __builtin_amdgcn_wmma_f32_16x16x32_bf16(
                false, pf.v, false, vf.v, (short)0, O[nt], false, false);
        }
        __syncthreads();   // protect ksh/vsh before next chunk's overwrite
    }

    // -- normalize + store: out[b, qbase+row, head*128 + d] --
    float linv[8];
#pragma unroll
    for (int e = 0; e < 8; ++e) linv[e] = __builtin_amdgcn_rcpf(lrow[e]);
    float* op = out + ((size_t)(b * T_SEQ + qbase)) * CMODEL + head * HD;
#pragma unroll
    for (int nt = 0; nt < 8; ++nt)
#pragma unroll
        for (int e = 0; e < 8; ++e) {
            int row = e + 8 * hh;
            op[(size_t)row * CMODEL + nt * 16 + r] = O[nt][e] * linv[e];
        }
}

// ---------------- launcher ----------------
extern "C" void kernel_launch(void* const* d_in, const int* in_sizes, int n_in,
                              void* d_out, int out_size, void* d_ws, size_t ws_size,
                              hipStream_t stream)
{
    const float* x  = (const float*)d_in[0];
    const float* Wq = (const float*)d_in[1];
    const float* Wk = (const float*)d_in[2];
    const float* Wv = (const float*)d_in[3];
    float* out = (float*)d_out;

    const int M = MROWS;                        // 4096
    const size_t nX  = (size_t)M * CMODEL;      // 8388608
    const size_t nWq = (size_t)CMODEL * CMODEL; // 4194304
    const size_t nWk = (size_t)DKV * CMODEL;    // 1048576
    const size_t nKV = (size_t)M * DKV;         // 2097152

    __bf16* xb  = (__bf16*)d_ws;
    __bf16* wqb = xb  + nX;
    __bf16* wkb = wqb + nWq;
    __bf16* wvb = wkb + nWk;
    __bf16* Qb  = wvb + nWk;
    __bf16* Kb  = Qb  + nX;
    __bf16* Vt  = Kb  + nKV;   // [512][4096] transposed

    // stage to bf16 (8 elems / thread)
    cvt_f32_to_bf16<<<(int)(nX  / 8 / 256), 256, 0, stream>>>(x,  xb,  (int)(nX  / 8));
    cvt_f32_to_bf16<<<(int)(nWq / 8 / 256), 256, 0, stream>>>(Wq, wqb, (int)(nWq / 8));
    cvt_f32_to_bf16<<<(int)(nWk / 8 / 256), 256, 0, stream>>>(Wk, wkb, (int)(nWk / 8));
    cvt_f32_to_bf16<<<(int)(nWk / 8 / 256), 256, 0, stream>>>(Wv, wvb, (int)(nWk / 8));

    // projections: y = x @ W^T  (64x64 per wave)
    dim3 blk(32, 1, 1);
    dim3 gq(M / 64, CMODEL / 64, 1);
    gemm_bf16_nt<<<gq, blk, 0, stream>>>(xb, wqb, Qb, M, CMODEL, CMODEL, 0);
    dim3 gkv(M / 64, DKV / 64, 1);
    gemm_bf16_nt<<<gkv, blk, 0, stream>>>(xb, wkb, Kb, M, DKV, CMODEL, 0);
    gemm_bf16_nt<<<gkv, blk, 0, stream>>>(xb, wvb, Vt, M, DKV, CMODEL, 1);

    // flash attention: 4-wave workgroup per 64-query tile per (b, head)
    dim3 ablk(128, 1, 1);
    dim3 ga(B_SZ * HEADS, T_SEQ / 64, 1);
    gqa_flash_attn<<<ga, ablk, 0, stream>>>(Qb, Kb, Vt, out);
}
